// Net_7834020348017
// MI455X (gfx1250) — compile-verified, matrix operand order checked by reading, not run on hardware
//
#include <hip/hip_runtime.h>
#include <hip/hip_bf16.h>

typedef __attribute__((ext_vector_type(16))) _Float16 v16h;
typedef __attribute__((ext_vector_type(8)))  _Float16 v8h;
typedef __attribute__((ext_vector_type(8)))  float    v8f;

#define DIMX 64

__device__ __forceinline__ v8f wmma_f16(v16h a, v16h b, v8f c) {
  return __builtin_amdgcn_wmma_f32_16x16x32_f16(false, a, false, b, (short)0, c, false, false);
}

// ---- A fragment from row-major LDS tile X[16][64] (f16): two contiguous b128 chunks.
// lane: M = lane&15, hi = lane>>4 ; element e -> K = (e&7) + 16*(e>>3) + 8*hi + 32*kb
__device__ __forceinline__ v16h frag_a(const _Float16* X, int mrow, int hi, int kb) {
  const _Float16* p = X + mrow * DIMX + (hi << 3) + (kb << 5);
  v8h lo = *(const v8h*)(p);
  v8h hi8 = *(const v8h*)(p + 16);
  return __builtin_shufflevector(lo, hi8, 0, 1, 2, 3, 4, 5, 6, 7, 8, 9, 10, 11, 12, 13, 14, 15);
}

// ---- Swizzled B weight storage: fragment-contiguous, one v16h (2x b128) per fragment.
// layout: dst[(((nb*2+kb)*32)+lane)*16 + e]
__device__ __forceinline__ void fill_bsw(_Float16* dst, const float* __restrict__ W,
                                         int tid, int Krows, int Ncols, int ldw) {
  for (int i = tid; i < 64 * 64; i += 256) {
    int e = i & 15, lane = (i >> 4) & 31, kb = (i >> 9) & 1, nb = i >> 10;
    int hi = lane >> 4;
    int k = (e & 7) + ((e >> 3) << 4) + (hi << 3) + (kb << 5);
    int n = nb * 16 + (lane & 15);
    dst[i] = (_Float16)((k < Krows && n < Ncols) ? W[k * ldw + n] : 0.f);
  }
}
__device__ __forceinline__ v16h frag_bsw(const _Float16* W, int lane, int kb, int nb) {
  return *(const v16h*)(W + ((((nb << 1) | kb) << 5) + lane) * 16);
}

// ---------------- initial node embeddings: out = (relu(f@W1+b1))@W2+b2, pad col63=0
__global__ __launch_bounds__(256) void k_init(const float* __restrict__ f,
                                              const float* __restrict__ W1, const float* __restrict__ b1,
                                              const float* __restrict__ W2, const float* __restrict__ b2,
                                              float* __restrict__ out, int N) {
  __shared__ float sH[4][64];
  int sub = threadIdx.x >> 6, j = threadIdx.x & 63;
  int node = blockIdx.x * 4 + sub;
  float f0 = 0.f, f1 = 0.f;
  if (node < N) { f0 = f[node * 2]; f1 = f[node * 2 + 1]; }
  float h = 0.f;
  if (j < 63) h = fmaxf(f0 * W1[j] + f1 * W1[63 + j] + b1[j], 0.f);
  sH[sub][j] = h;
  __syncthreads();
  float o = 0.f;
  if (j < 63) {
    o = b2[j];
    for (int i = 0; i < 63; ++i) o += sH[sub][i] * W2[i * 63 + j];
  }
  if (node < N) out[node * DIMX + j] = o;   // j==63 -> 0
}

__global__ void k_zero(float* __restrict__ p, long long n) {
  long long t = (long long)blockIdx.x * blockDim.x + threadIdx.x;
  for (; t < n; t += (long long)gridDim.x * blockDim.x) p[t] = 0.f;
}

__global__ void k_deg(const int* __restrict__ src, float* __restrict__ deg, int E) {
  int t = blockIdx.x * blockDim.x + threadIdx.x;
  if (t < E) atomicAdd(&deg[src[t]], 1.f);
}

// ---------------- per-node message kernel. MODE 0 = v2c (computes h2v & hv), MODE 1 = c2v.
template <int MODE>
__global__ __launch_bounds__(256) void k_msg(const float* __restrict__ x, const float* __restrict__ ef,
                                             const float* __restrict__ deg,
                                             const float* __restrict__ W1m, const float* __restrict__ b1m,
                                             const float* __restrict__ W2m, const float* __restrict__ b2m,
                                             const float* __restrict__ W1h, const float* __restrict__ b1h,
                                             const float* __restrict__ W2h, const float* __restrict__ b2h,
                                             float* __restrict__ msg, float* __restrict__ hvOut, int N) {
  __shared__ __align__(32) _Float16 sW1m[DIMX * DIMX];
  __shared__ __align__(32) _Float16 sW2m[DIMX * DIMX];
  __shared__ __align__(32) _Float16 sW1h[DIMX * DIMX];
  __shared__ float sB1m[64], sB2m[64], sB1h[64], sW2hv[64];
  __shared__ __align__(32) _Float16 sX[8][16 * DIMX];
  __shared__ __align__(32) _Float16 sH[8][16 * DIMX];
  __shared__ float sCE[8][16], sNORM[8][16], sHV[8][16];

  const int tid = threadIdx.x, w = tid >> 5, lane = tid & 31;
  fill_bsw(sW1m, W1m, tid, 64, 63, 63);
  fill_bsw(sW2m, W2m, tid, 63, 63, 63);
  if (MODE == 0) fill_bsw(sW1h, W1h, tid, 64, 63, 63);
  if (tid < 64) {
    sB1m[tid] = (tid < 63) ? b1m[tid] : 0.f;
    sB2m[tid] = (tid < 63) ? b2m[tid] : 0.f;
    if (MODE == 0) {
      sB1h[tid]  = (tid < 63) ? b1h[tid] : 0.f;
      sW2hv[tid] = (tid < 63) ? W2h[tid] : 0.f;
    }
  }
  const int n0 = blockIdx.x * 128 + w * 16;
  for (int i = lane; i < 16 * DIMX; i += 32) {
    int m = i >> 6, c = i & 63, node = n0 + m;
    sX[w][i] = (_Float16)((node < N) ? x[node * DIMX + c] : 0.f);
  }
  if (lane < 16) {
    int node = n0 + lane;
    sCE[w][lane] = (node < N) ? ef[node] : 0.f;
    float d = (node < N) ? deg[node] : 1.f;
    sNORM[w][lane] = (d > 0.f) ? 1.f / d : 0.f;   // deg==0 nodes are never sources
  }
  __syncthreads();

  const int mrow = lane & 15, hi = lane >> 4;
  v16h x0 = frag_a(&sX[w][0], mrow, hi, 0);
  v16h x1 = frag_a(&sX[w][0], mrow, hi, 1);

  if (MODE == 0) {
    // hidden = sigmoid(X @ W1h + b1h) ; hv = hidden @ W2h + b2h
#pragma unroll
    for (int nb = 0; nb < 4; ++nb) {
      int n = nb * 16 + mrow;
      v8f acc = {};
      acc = wmma_f16(x0, frag_bsw(sW1h, lane, 0, nb), acc);
      acc = wmma_f16(x1, frag_bsw(sW1h, lane, 1, nb), acc);
#pragma unroll
      for (int r = 0; r < 8; ++r) {
        int m = r + 8 * hi;
        float v = acc[r] + sB1h[n];
        sH[w][m * DIMX + n] = (_Float16)(1.f / (1.f + __expf(-v)));
      }
    }
    if (lane < 16) {
      float s = b2h[0];
      const v16h* row = (const v16h*)&sH[w][lane * DIMX];
#pragma unroll
      for (int c16 = 0; c16 < 4; ++c16) {
        v16h hh = row[c16];
#pragma unroll
        for (int e = 0; e < 16; ++e) s += (float)hh[e] * sW2hv[c16 * 16 + e];
      }
      sHV[w][lane] = s;
      int node = n0 + lane;
      if (node < N) hvOut[node] = s;
    }
  }

  // mlp2: H2 = relu((ce*X)@W1m + b1m); OUT = H2@W2m + b2m; msg[:, :63] = norm*OUT
  {
    _Float16 ch = (_Float16)sCE[w][mrow];
    v16h a0, a1;
#pragma unroll
    for (int e = 0; e < 16; ++e) { a0[e] = x0[e] * ch; a1[e] = x1[e] * ch; }
#pragma unroll
    for (int nb = 0; nb < 4; ++nb) {
      int n = nb * 16 + mrow;
      v8f acc = {};
      acc = wmma_f16(a0, frag_bsw(sW1m, lane, 0, nb), acc);
      acc = wmma_f16(a1, frag_bsw(sW1m, lane, 1, nb), acc);
#pragma unroll
      for (int r = 0; r < 8; ++r) {
        int m = r + 8 * hi;
        sH[w][m * DIMX + n] = (_Float16)fmaxf(acc[r] + sB1m[n], 0.f);
      }
    }
    v16h h0 = frag_a(&sH[w][0], mrow, hi, 0);
    v16h h1 = frag_a(&sH[w][0], mrow, hi, 1);
#pragma unroll
    for (int nb = 0; nb < 4; ++nb) {
      int n = nb * 16 + mrow;
      v8f acc = {};
      acc = wmma_f16(h0, frag_bsw(sW2m, lane, 0, nb), acc);
      acc = wmma_f16(h1, frag_bsw(sW2m, lane, 1, nb), acc);
#pragma unroll
      for (int r = 0; r < 8; ++r) {
        int m = r + 8 * hi, node = n0 + m;
        if (n < 63 && node < N) msg[node * DIMX + n] = sNORM[w][m] * (acc[r] + sB2m[n]);
      }
    }
    if (lane < 16) {
      int node = n0 + lane;
      if (node < N) {
        if (MODE == 0) msg[node * DIMX + 63] = sHV[w][lane] * sCE[w][lane];
        else           msg[node * DIMX + 63] = sNORM[w][lane] * x[node * DIMX + 63] * sCE[w][lane];
      }
    }
  }
}

// ---------------- scatter: aggr[dst] += msg[src], 1 thread per (edge, col)
__global__ __launch_bounds__(256) void k_scatter(const int* __restrict__ src, const int* __restrict__ dst,
                                                 const float* __restrict__ msg, float* __restrict__ aggr,
                                                 long long total) {
  long long t = (long long)blockIdx.x * blockDim.x + threadIdx.x;
  if (t >= total) return;
  int e = (int)(t >> 6), j = (int)(t & 63);
  atomicAdd(&aggr[(long long)dst[e] * DIMX + j], msg[(long long)src[e] * DIMX + j]);
}

// ---------------- node update. VAR=0: con (last = aggr63 - rhs); VAR=1: var (last = hv*aggr63)
template <int VAR>
__global__ __launch_bounds__(256) void k_update(float* __restrict__ xio, const float* __restrict__ aggr,
                                                const float* __restrict__ root, const float* __restrict__ bias,
                                                const float* __restrict__ rhs, const float* __restrict__ hv, int N) {
  __shared__ __align__(32) _Float16 sR[DIMX * DIMX];
  __shared__ float sB[64];
  __shared__ __align__(32) _Float16 sX[8][16 * DIMX];
  const int tid = threadIdx.x, w = tid >> 5, lane = tid & 31;
  fill_bsw(sR, root, tid, 64, 63, 63);
  if (tid < 64) sB[tid] = (tid < 63) ? bias[tid] : 0.f;
  const int n0 = blockIdx.x * 128 + w * 16;
  for (int i = lane; i < 16 * DIMX; i += 32) {
    int m = i >> 6, c = i & 63, node = n0 + m;
    sX[w][i] = (_Float16)((node < N) ? xio[node * DIMX + c] : 0.f);
  }
  __syncthreads();
  const int mrow = lane & 15, hi = lane >> 4;
  v16h a0 = frag_a(&sX[w][0], mrow, hi, 0);
  v16h a1 = frag_a(&sX[w][0], mrow, hi, 1);
#pragma unroll
  for (int nb = 0; nb < 4; ++nb) {
    int n = nb * 16 + mrow;
    v8f acc = {};
    acc = wmma_f16(a0, frag_bsw(sR, lane, 0, nb), acc);
    acc = wmma_f16(a1, frag_bsw(sR, lane, 1, nb), acc);
#pragma unroll
    for (int r = 0; r < 8; ++r) {
      int m = r + 8 * hi, node = n0 + m;
      if (node < N && n < 63)
        xio[node * DIMX + n] = fmaxf(acc[r] + sB[n] + aggr[node * DIMX + n], 0.f);
    }
  }
  if (lane < 16) {
    int node = n0 + lane;
    if (node < N) {
      if (VAR) xio[node * DIMX + 63] = hv[node] * aggr[node * DIMX + 63];
      else     xio[node * DIMX + 63] = aggr[node * DIMX + 63] - rhs[node];
    }
  }
}

// ---------------- head: 5x relu(x@W+b) then fc6 + log_softmax
__global__ __launch_bounds__(256) void k_head(const float* __restrict__ x,
                                              const float* W1, const float* B1, const float* W2, const float* B2,
                                              const float* W3, const float* B3, const float* W4, const float* B4,
                                              const float* W5, const float* B5, const float* W6, const float* B6,
                                              float* __restrict__ out, int N) {
  __shared__ __align__(32) _Float16 sW[DIMX * DIMX];
  __shared__ float sB[64];
  __shared__ __align__(32) _Float16 sBufA[8][16 * DIMX];
  __shared__ __align__(32) _Float16 sBufB[8][16 * DIMX];
  __shared__ float sW6[128], sB6[2];
  const int tid = threadIdx.x, w = tid >> 5, lane = tid & 31;
  if (tid < 128) sW6[tid] = W6[tid];
  if (tid < 2)  sB6[tid] = B6[tid];
  const int n0 = blockIdx.x * 128 + w * 16;
  for (int i = lane; i < 16 * DIMX; i += 32) {
    int m = i >> 6, c = i & 63, node = n0 + m;
    sBufA[w][i] = (_Float16)((node < N) ? x[node * DIMX + c] : 0.f);
  }
  const float* Ws[5] = {W1, W2, W3, W4, W5};
  const float* Bs[5] = {B1, B2, B3, B4, B5};
  _Float16* cur = &sBufA[w][0];
  _Float16* nxt = &sBufB[w][0];
  const int mrow = lane & 15, hi = lane >> 4;
  for (int l = 0; l < 5; ++l) {
    __syncthreads();
    fill_bsw(sW, Ws[l], tid, 64, 64, 64);
    if (tid < 64) sB[tid] = Bs[l][tid];
    __syncthreads();
    v16h a0 = frag_a(cur, mrow, hi, 0);
    v16h a1 = frag_a(cur, mrow, hi, 1);
#pragma unroll
    for (int nb = 0; nb < 4; ++nb) {
      int n = nb * 16 + mrow;
      v8f acc = {};
      acc = wmma_f16(a0, frag_bsw(sW, lane, 0, nb), acc);
      acc = wmma_f16(a1, frag_bsw(sW, lane, 1, nb), acc);
#pragma unroll
      for (int r = 0; r < 8; ++r) {
        int m = r + 8 * hi;
        nxt[m * DIMX + n] = (_Float16)fmaxf(acc[r] + sB[n], 0.f);
      }
    }
    _Float16* t = cur; cur = nxt; nxt = t;
  }
  if (lane < 16) {
    int node = n0 + lane;
    if (node < N) {
      float a = sB6[0], b = sB6[1];
      const v16h* row = (const v16h*)&cur[lane * DIMX];
#pragma unroll
      for (int c16 = 0; c16 < 4; ++c16) {
        v16h hh = row[c16];
#pragma unroll
        for (int e = 0; e < 16; ++e) {
          float xv = (float)hh[e];
          a += xv * sW6[2 * (c16 * 16 + e)];
          b += xv * sW6[2 * (c16 * 16 + e) + 1];
        }
      }
      float mm = fmaxf(a, b);
      float lse = mm + __logf(__expf(a - mm) + __expf(b - mm));
      out[node * 2]     = a - lse;
      out[node * 2 + 1] = b - lse;
    }
  }
}

extern "C" void kernel_launch(void* const* d_in, const int* in_sizes, int n_in,
                              void* d_out, int out_size, void* d_ws, size_t ws_size,
                              hipStream_t stream) {
  const int Nv = in_sizes[0] / 2;
  const int Nc = in_sizes[1] / 2;
  const int E  = in_sizes[2] / 2;

  const float* vf  = (const float*)d_in[0];
  const float* cf  = (const float*)d_in[1];
  const int*   eiv = (const int*)d_in[2];
  const float* efv = (const float*)d_in[3];
  const int*   eic = (const int*)d_in[4];
  const float* efc = (const float*)d_in[5];
  const float* rhs = (const float*)d_in[6];

  // params flattened as JAX pytree (alphabetical dict keys):
  // c2v_1..4 {bias,mlp.W1,mlp.W2,mlp.b1,mlp.b2,root}, con_mlp {W1,W2,b1,b2},
  // fc1..fc6 {W,b}, h2v_1..4 {W1,W2,b1,b2}, v2c_1..4 (as c2v), var_mlp {W1,W2,b1,b2}
  auto P = [&](int i) { return (const float*)d_in[i]; };
  const int C2V = 8, CON = 32, FC = 36, H2V = 48, V2C = 64, VARM = 88;

  float* ws = (float*)d_ws;
  const int M = (Nv > Nc) ? Nv : Nc;
  float* vars = ws;
  float* cons = vars + (size_t)Nv * DIMX;
  float* msg  = cons + (size_t)Nc * DIMX;
  float* aggr = msg  + (size_t)M * DIMX;
  float* hv   = aggr + (size_t)M * DIMX;
  float* degV = hv + Nv;
  float* degC = degV + Nv;

  dim3 blk(256);
  const int gTileV = (Nv + 127) / 128, gTileC = (Nc + 127) / 128;
  const long long scat = (long long)E * DIMX;
  const int gScat = (int)((scat + 255) / 256);

  // initial embeddings (note reference swaps naming: con_mlp -> var nodes)
  k_init<<<(Nv + 3) / 4, blk, 0, stream>>>(vf, P(CON), P(CON + 2), P(CON + 1), P(CON + 3), vars, Nv);
  k_init<<<(Nc + 3) / 4, blk, 0, stream>>>(cf, P(VARM), P(VARM + 2), P(VARM + 1), P(VARM + 3), cons, Nc);

  // degrees (src side of each edge list)
  k_zero<<<256, blk, 0, stream>>>(degV, Nv);
  k_zero<<<256, blk, 0, stream>>>(degC, Nc);
  k_deg<<<(E + 255) / 256, blk, 0, stream>>>(eiv, degV, E);
  k_deg<<<(E + 255) / 256, blk, 0, stream>>>(eic, degC, E);

  for (int k = 0; k < 4; ++k) {
    const int iv = V2C + k * 6;   // bias, W1, W2, b1, b2, root
    const int ic = C2V + k * 6;
    const int ih = H2V + k * 4;   // W1, W2, b1, b2

    // v2c: per-var-node messages (incl. hv), scatter to con nodes, con update
    k_msg<0><<<gTileV, blk, 0, stream>>>(vars, efv, degV,
                                         P(iv + 1), P(iv + 3), P(iv + 2), P(iv + 4),
                                         P(ih), P(ih + 2), P(ih + 1), P(ih + 3),
                                         msg, hv, Nv);
    k_zero<<<256, blk, 0, stream>>>(aggr, (long long)Nc * DIMX);
    k_scatter<<<gScat, blk, 0, stream>>>(eiv, eiv + E, msg, aggr, scat);
    k_update<0><<<gTileC, blk, 0, stream>>>(cons, aggr, P(iv + 5), P(iv), rhs, nullptr, Nc);

    // c2v: per-con-node messages, scatter to var nodes, var update (uses hv)
    k_msg<1><<<gTileC, blk, 0, stream>>>(cons, efc, degC,
                                         P(ic + 1), P(ic + 3), P(ic + 2), P(ic + 4),
                                         P(ic + 1), P(ic + 3), P(ic + 2), P(ic + 4),
                                         msg, nullptr, Nc);
    k_zero<<<256, blk, 0, stream>>>(aggr, (long long)Nv * DIMX);
    k_scatter<<<gScat, blk, 0, stream>>>(eic, eic + E, msg, aggr, scat);
    k_update<1><<<gTileV, blk, 0, stream>>>(vars, aggr, P(ic + 5), P(ic), nullptr, hv, Nv);
  }

  // head: fc1..fc5 (relu) + fc6 + log_softmax
  k_head<<<gTileV, blk, 0, stream>>>(vars,
                                     P(FC + 0), P(FC + 1), P(FC + 2), P(FC + 3),
                                     P(FC + 4), P(FC + 5), P(FC + 6), P(FC + 7),
                                     P(FC + 8), P(FC + 9), P(FC + 10), P(FC + 11),
                                     (float*)d_out, Nv);
}
